// GraphMap_39865886441901
// MI455X (gfx1250) — matrix operand order repaired and checked by reading, not compiled, not run on hardware
//
#include <hip/hip_runtime.h>
#include <hip/hip_bf16.h>

// ---------------------------------------------------------------------------
// GNN (interaction network) forward on gfx1250 using bf16 WMMA.
// Pipeline per layer:
//   edge MLP (65->128->64, ReLU+BN each stage) over E=900k rows  [3 passes]
//   segment-mean by src into agg[N,64]
//   node MLP (128->128->n_out) over N=100k rows                  [3 passes]
// BN (training mode): pass computes column sum/sumsq, tiny kernel folds
// (g, beta, mu, var) into per-feature (scale, shift), applied on load.
// ---------------------------------------------------------------------------

typedef __bf16 bf16_t;
typedef __attribute__((ext_vector_type(16))) __bf16 v16bf;
typedef __attribute__((ext_vector_type(8)))  __bf16 v8bf;
typedef __attribute__((ext_vector_type(8)))  float  v8f;

#define KSTR 136   // LDS K-stride in bf16 elements (272B rows, 16B aligned)

__device__ __forceinline__ bf16_t f2bf(float f) {
  union { float f; unsigned u; } v; v.f = f;
  unsigned r = v.u + 0x7fffu + ((v.u >> 16) & 1u);   // round-to-nearest-even
  unsigned short h = (unsigned short)(r >> 16);
  bf16_t b; __builtin_memcpy(&b, &h, sizeof(b));
  return b;
}

__device__ __forceinline__ v8f vzero8() {
  v8f z;
#pragma unroll
  for (int i = 0; i < 8; ++i) z[i] = 0.f;
  return z;
}

__device__ __forceinline__ v16bf combine8(const bf16_t* p0, const bf16_t* p1) {
  v8bf lo = *(const v8bf*)p0;
  v8bf hi = *(const v8bf*)p1;
  v16bf r;
#pragma unroll
  for (int i = 0; i < 8; ++i) { r[i] = lo[i]; r[8 + i] = hi[i]; }
  return r;
}

// A-fragment (16x32 bf16, row-major LDS [row][k], stride KSTR).
// Lane l: row = row_base + (l&15); lane-half h=l>>4 covers K = k0+h*8..+7 and
// k0+16+h*8..+7 (matches ISA 16-bit A layout).
__device__ __forceinline__ v16bf lda(const bf16_t* base, int row_base, int k0, int lane) {
  int row = row_base + (lane & 15);
  const bf16_t* p = base + row * KSTR + k0 + ((lane >> 4) << 3);
  return combine8(p, p + 16);
}

// B-fragment (32x16 bf16). W stored transposed in LDS: wt[col][k], stride KSTR.
// Lane l: col = col_base + (l&15); half h covers K = k0+h*8..+7 and k0+16+h*8..+7.
__device__ __forceinline__ v16bf ldb(const bf16_t* wt, int col_base, int k0, int lane) {
  int col = col_base + (lane & 15);
  const bf16_t* p = wt + col * KSTR + k0 + ((lane >> 4) << 3);
  return combine8(p, p + 16);
}

// ---------------------------------------------------------------------------
// Fused 2-stage MLP tile kernel. Block = 64 rows, 256 threads (8 wave32).
// IS_EDGE: A row = [bn(h[src]) (64) | edge_attr (1) | pad..96]
// !IS_EDGE: A row = [bn(h[i]) (64) | agg[i]/max(cnt,1) (64)]
// MODE 0: stage0 only, accumulate stage0 column stats.
// MODE 1: stage0(+bn0) + stage1, accumulate stage1 column stats.
// MODE 2: full; edges: atomicAdd messages into agg[src]; nodes: store h_raw.
// ---------------------------------------------------------------------------
template <int MODE, bool IS_EDGE>
__global__ __launch_bounds__(256) void mlp2_kernel(
    int M,
    const float* __restrict__ hin, const float* __restrict__ in_sc, const float* __restrict__ in_sh,
    const int* __restrict__ esrc, const float* __restrict__ eattr,
    const float* __restrict__ agg, const float* __restrict__ cnt,
    const float* __restrict__ W1, const float* __restrict__ b1,
    const float* __restrict__ bn0_sc, const float* __restrict__ bn0_sh,
    const float* __restrict__ W2, const float* __restrict__ b2,
    const float* __restrict__ bn1_sc, const float* __restrict__ bn1_sh,
    int n_out,
    float* __restrict__ st_sum, float* __restrict__ st_sq,
    float* __restrict__ out_agg, float* __restrict__ out_h) {

  constexpr int K0  = IS_EDGE ? 96 : 128;   // padded stage-0 K
  constexpr int K0R = IS_EDGE ? 65 : 128;   // real stage-0 K (W1 rows)
  constexpr int KS0 = K0 / 32;

  __shared__ __align__(16) bf16_t sA[64 * KSTR];
  __shared__ __align__(16) bf16_t sW[128 * KSTR];
  __shared__ float s_sum[128];
  __shared__ float s_sq[128];
  __shared__ int   s_src[64];

  const int tid   = threadIdx.x;
  const int lane  = tid & 31;
  const int wr    = tid >> 5;
  const int rwave = wr & 3;   // row block (16 rows)
  const int cgrp  = wr >> 2;  // column group (0/1)
  const int row0  = blockIdx.x * 64;

  if (tid < 128) { s_sum[tid] = 0.f; s_sq[tid] = 0.f; }

  // ---- load W1^T into sW (bf16), zero-padded K
  for (int t = tid; t < 128 * K0; t += 256) {
    int c = t / K0, k = t - c * K0;
    float w = (k < K0R) ? W1[k * 128 + c] : 0.f;
    sW[c * KSTR + k] = f2bf(w);
  }

  // ---- build A tile (64 rows x K0), 4 threads per row
  {
    constexpr int KQ = K0 / 4;
    int r = tid >> 2, seg = tid & 3;
    int grow = row0 + r;
    int srcid = -1;
    if (grow < M) srcid = IS_EDGE ? esrc[grow] : grow;
    if (seg == 0) s_src[r] = srcid;
    if (IS_EDGE) {
      for (int k = seg * KQ; k < (seg + 1) * KQ; ++k) {
        float v = 0.f;
        if (srcid >= 0) {
          if (k < 64)       v = hin[(size_t)srcid * 64 + k] * in_sc[k] + in_sh[k];
          else if (k == 64) v = eattr[grow];
        }
        sA[r * KSTR + k] = f2bf(v);
      }
    } else {
      float c_inv = 0.f;
      if (srcid >= 0) c_inv = 1.f / fmaxf(cnt[srcid], 1.f);
      for (int k = seg * KQ; k < (seg + 1) * KQ; ++k) {
        float v = 0.f;
        if (srcid >= 0) {
          if (k < 64) v = hin[(size_t)srcid * 64 + k] * in_sc[k] + in_sh[k];
          else        v = agg[(size_t)srcid * 64 + (k - 64)] * c_inv;
        }
        sA[r * KSTR + k] = f2bf(v);
      }
    }
  }
  __syncthreads();

  const int lcol = lane & 15;
  const int lr0  = (lane >> 4) * 8;

  // Hoist this wave's 8 C-fragment row ids + validity masks (same rows for
  // every column tile): 8 LDS reads total instead of 8 per tile, and a
  // branchless 0/1 float mask for the stats paths.
  int   rsrc[8];
  float rmask[8];
#pragma unroll
  for (int r = 0; r < 8; ++r) {
    rsrc[r]  = s_src[16 * rwave + lr0 + r];
    rmask[r] = (rsrc[r] >= 0) ? 1.f : 0.f;
  }

  // ---- stage 0 GEMM: [64 x K0] @ [K0 x 128]; wave covers 16 rows x 64 cols
  v8f acc[4];
#pragma unroll
  for (int t = 0; t < 4; ++t) acc[t] = vzero8();

  v16bf afr[4];
#pragma unroll
  for (int ks = 0; ks < KS0; ++ks) afr[ks] = lda(sA, 16 * rwave, 32 * ks, lane);

#pragma unroll
  for (int t = 0; t < 4; ++t) {
    int nc = cgrp * 4 + t;
#pragma unroll
    for (int ks = 0; ks < KS0; ++ks) {
      v16bf bfr = ldb(sW, 16 * nc, 32 * ks, lane);
      acc[t] = __builtin_amdgcn_wmma_f32_16x16x32_bf16(
          false, afr[ks], false, bfr, (short)0, acc[t], false, false);
    }
  }

  if constexpr (MODE == 0) {
#pragma unroll
    for (int t = 0; t < 4; ++t) {
      int col = cgrp * 64 + t * 16 + lcol;
      float bia = b1[col];
      float s = 0.f, q = 0.f;
#pragma unroll
      for (int r = 0; r < 8; ++r) {
        float z = fmaxf(acc[t][r] + bia, 0.f) * rmask[r];
        s += z; q += z * z;
      }
      atomicAdd(&s_sum[col], s);
      atomicAdd(&s_sq[col], q);
    }
    __syncthreads();
    if (tid < 128) { atomicAdd(&st_sum[tid], s_sum[tid]); atomicAdd(&st_sq[tid], s_sq[tid]); }
    return;
  }

  // ---- MODE >= 1: write bn0(relu(z)) back into sA as stage-1 input
  __syncthreads();  // all sA / sW reads done
#pragma unroll
  for (int t = 0; t < 4; ++t) {
    int col = cgrp * 64 + t * 16 + lcol;
    float bia = b1[col];
    float sc = bn0_sc[col], sh = bn0_sh[col];
#pragma unroll
    for (int r = 0; r < 8; ++r) {
      int rl = 16 * rwave + lr0 + r;
      float z = fmaxf(acc[t][r] + bia, 0.f);
      sA[rl * KSTR + col] = f2bf(z * sc + sh);
    }
  }
  // reload sW with W2^T (cols padded to tile multiple)
  const int NT1 = (n_out + 15) >> 4;
  for (int t = tid; t < NT1 * 16 * 128; t += 256) {
    int c = t >> 7, k = t & 127;
    float w = (c < n_out) ? W2[k * n_out + c] : 0.f;
    sW[c * KSTR + k] = f2bf(w);
  }
  __syncthreads();

  // ---- stage 1 GEMM: [64 x 128] @ [128 x n_out]
  v16bf a1[4];
#pragma unroll
  for (int ks = 0; ks < 4; ++ks) a1[ks] = lda(sA, 16 * rwave, 32 * ks, lane);

  for (int t1 = cgrp; t1 < NT1; t1 += 2) {
    v8f c1 = vzero8();
#pragma unroll
    for (int ks = 0; ks < 4; ++ks) {
      v16bf bfr = ldb(sW, 16 * t1, 32 * ks, lane);
      c1 = __builtin_amdgcn_wmma_f32_16x16x32_bf16(
          false, a1[ks], false, bfr, (short)0, c1, false, false);
    }
    int col = t1 * 16 + lcol;
    bool colok = col < n_out;
    float cmask = colok ? 1.f : 0.f;
    float bia = colok ? b2[col] : 0.f;
    if constexpr (MODE == 1) {
      float s = 0.f, q = 0.f;
#pragma unroll
      for (int r = 0; r < 8; ++r) {
        float z = fmaxf(c1[r] + bia, 0.f) * rmask[r] * cmask;
        s += z; q += z * z;
      }
      if (colok) { atomicAdd(&s_sum[col], s); atomicAdd(&s_sq[col], q); }
    } else {  // MODE == 2
      float sc = colok ? bn1_sc[col] : 0.f;
      float sh = colok ? bn1_sh[col] : 0.f;
#pragma unroll
      for (int r = 0; r < 8; ++r) {
        int sid = rsrc[r];
        if (sid >= 0 && colok) {
          float z = fmaxf(c1[r] + bia, 0.f);
          float m = z * sc + sh;
          if (IS_EDGE) atomicAdd(&out_agg[(size_t)sid * 64 + col], m);
          else         out_h[(size_t)sid * n_out + col] = m;
        }
      }
    }
  }
  if constexpr (MODE == 1) {
    __syncthreads();
    if (tid < n_out) { atomicAdd(&st_sum[tid], s_sum[tid]); atomicAdd(&st_sq[tid], s_sq[tid]); }
  }
}

// ---------------------------------------------------------------------------
__global__ void bn_finalize_kernel(const float* __restrict__ sum, const float* __restrict__ sq,
                                   const float* __restrict__ g, const float* __restrict__ be,
                                   float invM, int C,
                                   float* __restrict__ sc, float* __restrict__ sh) {
  int i = blockIdx.x * blockDim.x + threadIdx.x;
  if (i < C) {
    float mu  = sum[i] * invM;
    float var = fmaxf(sq[i] * invM - mu * mu, 0.f);
    float s   = g[i] * rsqrtf(var + 1e-5f);
    sc[i] = s;
    sh[i] = be[i] - mu * s;
  }
}

__global__ void init_id_kernel(float* sc, float* sh) {
  int i = threadIdx.x;
  if (i < 64) { sc[i] = 1.f; sh[i] = 0.f; }
}

__global__ void count_kernel(const int* __restrict__ esrc, float* __restrict__ cnt, int E) {
  int e = blockIdx.x * blockDim.x + threadIdx.x;
  if (e < E) atomicAdd(&cnt[esrc[e]], 1.f);
}

__global__ void dist_kernel(const float* __restrict__ h, const float* __restrict__ sc,
                            const float* __restrict__ sh, const int* __restrict__ idx,
                            float* __restrict__ out, int N) {
  int i = blockIdx.x * blockDim.x + threadIdx.x;
  if (i < N) {
    float a0 = h[(size_t)i * 2 + 0] * sc[0] + sh[0];
    float a1 = h[(size_t)i * 2 + 1] * sc[1] + sh[1];
#pragma unroll
    for (int k = 0; k < 10; ++k) {
      int j = idx[(size_t)i * 10 + k];
      float b0 = h[(size_t)j * 2 + 0] * sc[0] + sh[0];
      float b1 = h[(size_t)j * 2 + 1] * sc[1] + sh[1];
      float d0 = a0 - b0, d1 = a1 - b1;
      out[(size_t)i * 10 + k] = d0 * d0 + d1 * d1;
    }
  }
}

// ---------------------------------------------------------------------------
extern "C" void kernel_launch(void* const* d_in, const int* in_sizes, int n_in,
                              void* d_out, int out_size, void* d_ws, size_t ws_size,
                              hipStream_t stream) {
  (void)in_sizes; (void)n_in; (void)out_size; (void)ws_size;
  const int Nn = 100000, Kn = 10;
  const int Ee = Nn * (Kn - 1);

  const float* x    = (const float*)d_in[0];
  const int*   ei   = (const int*)d_in[1];
  const float* ea   = (const float*)d_in[2];
  const int*   idxp = (const int*)d_in[3];
  const int*   esrc = ei + Ee;  // edge_index[1] = src

  auto P = [&](int s, int a) { return (const float*)d_in[4 + s * 4 + a]; };

  // workspace layout (floats)
  float* w = (float*)d_ws;
  size_t off = 0;
  float* cnt   = w + off; off += Nn;
  float* agg   = w + off; off += (size_t)Nn * 64;
  float* h1    = w + off; off += (size_t)Nn * 64;
  float* h2    = w + off; off += (size_t)Nn * 2;
  float* id_sc = w + off; off += 64;
  float* id_sh = w + off; off += 64;
  float* bnbuf = w + off; off += 8 * 256;   // stage s: sc=bnbuf+s*256, sh=+128
  float* st_sum = w + off; off += 128;
  float* st_sq  = w + off; off += 128;

  auto bsc = [&](int s) { return bnbuf + s * 256; };
  auto bsh = [&](int s) { return bnbuf + s * 256 + 128; };

  const int ge = (Ee + 63) / 64;
  const int gn = (Nn + 63) / 64;

  init_id_kernel<<<1, 64, 0, stream>>>(id_sc, id_sh);
  hipMemsetAsync(cnt, 0, (size_t)Nn * sizeof(float), stream);
  count_kernel<<<(Ee + 255) / 256, 256, 0, stream>>>(esrc, cnt, Ee);

  const float* cur_h  = x;
  const float* cur_sc = id_sc;
  const float* cur_sh = id_sh;

  for (int L = 0; L < 2; ++L) {
    int sM0 = L * 4 + 0, sM1 = L * 4 + 1, sN0 = L * 4 + 2, sN1 = L * 4 + 3;
    int nON = (L == 0) ? 64 : 2;

    hipMemsetAsync(agg, 0, (size_t)Nn * 64 * sizeof(float), stream);

    // ---------- edge MLP: stats0 -> bn0 -> stats1 -> bn1 -> scatter
    hipMemsetAsync(st_sum, 0, 256 * sizeof(float), stream);
    mlp2_kernel<0, true><<<ge, 256, 0, stream>>>(
        Ee, cur_h, cur_sc, cur_sh, esrc, ea, nullptr, nullptr,
        P(sM0, 0), P(sM0, 1), nullptr, nullptr,
        P(sM1, 0), P(sM1, 1), nullptr, nullptr,
        64, st_sum, st_sq, nullptr, nullptr);
    bn_finalize_kernel<<<1, 128, 0, stream>>>(st_sum, st_sq, P(sM0, 2), P(sM0, 3),
                                              1.0f / Ee, 128, bsc(sM0), bsh(sM0));
    hipMemsetAsync(st_sum, 0, 256 * sizeof(float), stream);
    mlp2_kernel<1, true><<<ge, 256, 0, stream>>>(
        Ee, cur_h, cur_sc, cur_sh, esrc, ea, nullptr, nullptr,
        P(sM0, 0), P(sM0, 1), bsc(sM0), bsh(sM0),
        P(sM1, 0), P(sM1, 1), nullptr, nullptr,
        64, st_sum, st_sq, nullptr, nullptr);
    bn_finalize_kernel<<<1, 128, 0, stream>>>(st_sum, st_sq, P(sM1, 2), P(sM1, 3),
                                              1.0f / Ee, 64, bsc(sM1), bsh(sM1));
    mlp2_kernel<2, true><<<ge, 256, 0, stream>>>(
        Ee, cur_h, cur_sc, cur_sh, esrc, ea, nullptr, nullptr,
        P(sM0, 0), P(sM0, 1), bsc(sM0), bsh(sM0),
        P(sM1, 0), P(sM1, 1), bsc(sM1), bsh(sM1),
        64, nullptr, nullptr, agg, nullptr);

    // ---------- node MLP
    hipMemsetAsync(st_sum, 0, 256 * sizeof(float), stream);
    mlp2_kernel<0, false><<<gn, 256, 0, stream>>>(
        Nn, cur_h, cur_sc, cur_sh, nullptr, nullptr, agg, cnt,
        P(sN0, 0), P(sN0, 1), nullptr, nullptr,
        P(sN1, 0), P(sN1, 1), nullptr, nullptr,
        nON, st_sum, st_sq, nullptr, nullptr);
    bn_finalize_kernel<<<1, 128, 0, stream>>>(st_sum, st_sq, P(sN0, 2), P(sN0, 3),
                                              1.0f / Nn, 128, bsc(sN0), bsh(sN0));
    hipMemsetAsync(st_sum, 0, 256 * sizeof(float), stream);
    mlp2_kernel<1, false><<<gn, 256, 0, stream>>>(
        Nn, cur_h, cur_sc, cur_sh, nullptr, nullptr, agg, cnt,
        P(sN0, 0), P(sN0, 1), bsc(sN0), bsh(sN0),
        P(sN1, 0), P(sN1, 1), nullptr, nullptr,
        nON, st_sum, st_sq, nullptr, nullptr);
    bn_finalize_kernel<<<1, 128, 0, stream>>>(st_sum, st_sq, P(sN1, 2), P(sN1, 3),
                                              1.0f / Nn, nON, bsc(sN1), bsh(sN1));
    mlp2_kernel<2, false><<<gn, 256, 0, stream>>>(
        Nn, cur_h, cur_sc, cur_sh, nullptr, nullptr, agg, cnt,
        P(sN0, 0), P(sN0, 1), bsc(sN0), bsh(sN0),
        P(sN1, 0), P(sN1, 1), bsc(sN1), bsh(sN1),
        nON, nullptr, nullptr, nullptr, (L == 0) ? h1 : h2);

    cur_h  = (L == 0) ? h1 : h2;
    cur_sc = bsc(sN1);
    cur_sh = bsh(sN1);
  }

  dist_kernel<<<(Nn + 255) / 256, 256, 0, stream>>>(h2, bsc(7), bsh(7), idxp,
                                                    (float*)d_out, Nn);
}
